// RelationMessagePassingBase_3212635537896
// MI455X (gfx1250) — matrix operand order repaired and checked by reading, not compiled, not run on hardware
//
#include <hip/hip_runtime.h>

// ---- CDNA5 WMMA types ------------------------------------------------------
typedef __attribute__((ext_vector_type(16))) __bf16 v16bf;
typedef __attribute__((ext_vector_type(8)))  __bf16 bf8v;
typedef __attribute__((ext_vector_type(4)))  __bf16 bf4v;
typedef __attribute__((ext_vector_type(8)))  float  v8f;

// Fast mish using only hardware TRANS ops (v_exp_f32, v_rcp_f32):
//   p = e^x;  e^softplus = 1+p;  tanh(sp) = ((1+p)^2-1)/((1+p)^2+1)
//   => mish(x) = x * n / (n+2),  n = p*(p+2)
// Guard x>15: tanh(sp) == 1 to 1e-13, and e^x would overflow fp32 near 88.
__device__ __forceinline__ float mishf(float x) {
    float p = __expf(x);
    float n = p * (p + 2.0f);
    float m = x * n * __builtin_amdgcn_rcpf(n + 2.0f);
    return (x > 15.0f) ? x : m;
}

// One wave computes a 16-row strip times NT2 16-col tiles of D = A*64,
// accumulating D/32 bf16 WMMA steps.  A comes from LDS (padded, ldb elems/row),
// B comes from global bf16 transposed weights Wt[n*D + k] (L2-resident).
template<int D, int NT2>
__device__ __forceinline__ void wave_gemm(const __bf16* Arows, int ldb,
                                          const __bf16* __restrict__ Wt,
                                          const float*  __restrict__ bias,
                                          int nBase, int lane, v8f (&acc)[NT2])
{
    const int nlane = lane & 15;
    const int lsel  = (lane >> 4) << 3;   // 0 for lanes 0-15, 8 for lanes 16-31

    // init accumulators with the (per-column) bias
#pragma unroll
    for (int t = 0; t < NT2; ++t) {
        float bv = bias[(nBase + t) * 16 + nlane];
#pragma unroll
        for (int j = 0; j < 8; ++j) acc[t][j] = bv;
    }

    for (int k0 = 0; k0 < D; k0 += 32) {
        // A fragment (16x32 bf16): lane&15 = row M; lanes 0-15 hold K k0+0..7 and
        // k0+16..23, lanes 16-31 hold K k0+8..15 and k0+24..31 (ISA 7.12.2).
        const __bf16* ap = Arows + nlane * ldb + k0 + lsel;
        bf8v alo = *(const bf8v*)(ap);
        bf8v ahi = *(const bf8v*)(ap + 16);
        v16bf af = __builtin_shufflevector(alo, ahi,
                     0,1,2,3,4,5,6,7,8,9,10,11,12,13,14,15);
#pragma unroll
        for (int t = 0; t < NT2; ++t) {
            // B fragment (32x16 bf16): lane&15 = col N; lanes 0-15: K k0+0..7 and
            // k0+16..23, lanes 16-31: K k0+8..15 and k0+24..31 (same order as A).
            const __bf16* bp = Wt + (size_t)((nBase + t) * 16 + nlane) * D
                                  + k0 + lsel;
            bf8v blo = *(const bf8v*)(bp);
            bf8v bhi = *(const bf8v*)(bp + 16);
            v16bf bfv = __builtin_shufflevector(blo, bhi,
                          0,1,2,3,4,5,6,7,8,9,10,11,12,13,14,15);
            acc[t] = __builtin_amdgcn_wmma_f32_16x16x32_bf16(
                         false, af, false, bfv, (short)0, acc[t], false, false);
        }
    }
}

// Fused per-relation kernel: gather -> GEMM1+bias -> mish -> GEMM2+bias ->
// residual add (fp32) -> store.  Tile = 32 fact-rows x D cols, 4 waves.
template<int A>
__global__ void __launch_bounds__(128)
rel_msg_kernel(const float* __restrict__ x, const int* __restrict__ idx,
               const __bf16* __restrict__ w1t, const float* __restrict__ b1,
               const __bf16* __restrict__ w2t, const float* __restrict__ b2,
               float* __restrict__ out, int nFacts)
{
    constexpr int D   = A * 64;
    constexpr int MT  = 32;          // fact rows per block
    constexpr int LDB = D + 8;       // bf16 row pitch (16B pad: bank-conflict break)
    constexpr int LDF = D + 4;       // fp32 row pitch
    constexpr int NT2 = 2 * A;       // 16-col tiles per wave (half of D/16)

    __shared__ float  hf[MT * LDF];  // fp32 gathered h (exact residual)
    __shared__ __bf16 hb[MT * LDB];  // bf16 h  (A of GEMM1)
    __shared__ __bf16 tb[MT * LDB];  // bf16 mish(h@w1+b1) (A of GEMM2)

    const int f0    = blockIdx.x * MT;
    const int tid   = threadIdx.x;
    const int lane  = tid & 31;
    const int wave  = tid >> 5;
    const int r0    = (wave >> 1) * 16;   // row strip
    const int nBase = (wave & 1) * NT2;   // column-tile half
    const int nlane = lane & 15;
    const int lsel  = (lane >> 4) << 3;

    // ---- Phase 1: gather x[idx] rows (float4), store fp32 + bf16 to LDS ----
    constexpr int VPR = D / 4;            // float4s per fact row
    for (int i = tid; i < MT * VPR; i += 128) {
        int row  = i / VPR;
        int col  = (i - row * VPR) * 4;
        int fact = f0 + row;
        if (fact >= nFacts) fact = nFacts - 1;     // clamp: extra rows unused
        int node = idx[fact * A + (col >> 6)];
        float4 v = *(const float4*)(x + (size_t)node * 64 + (col & 63));
        *(float4*)&hf[row * LDF + col] = v;
        bf4v h4;
        h4[0] = (__bf16)v.x; h4[1] = (__bf16)v.y;
        h4[2] = (__bf16)v.z; h4[3] = (__bf16)v.w;
        *(bf4v*)&hb[row * LDB + col] = h4;
    }
    __syncthreads();

    v8f acc[NT2];

    // ---- Phase 2: t = mish(h @ w1 + b1) -----------------------------------
    wave_gemm<D, NT2>(&hb[r0 * LDB], LDB, w1t, b1, nBase, lane, acc);
#pragma unroll
    for (int t = 0; t < NT2; ++t)
#pragma unroll
        for (int j = 0; j < 8; ++j) {
            int r = r0 + j + lsel;                    // C layout: M = j (+8 hi half)
            int c = (nBase + t) * 16 + nlane;
            tb[r * LDB + c] = (__bf16)mishf(acc[t][j]);
        }
    __syncthreads();

    // ---- Phase 3: out = h + (t @ w2 + b2) ---------------------------------
    wave_gemm<D, NT2>(&tb[r0 * LDB], LDB, w2t, b2, nBase, lane, acc);
#pragma unroll
    for (int t = 0; t < NT2; ++t)
#pragma unroll
        for (int j = 0; j < 8; ++j) {
            int r    = r0 + j + lsel;
            int c    = (nBase + t) * 16 + nlane;
            float vv = acc[t][j] + hf[r * LDF + c];   // exact fp32 residual
            int fact = f0 + r;
            if (fact < nFacts) out[(size_t)fact * D + c] = vv;
        }
}

// Pre-pass: wT_bf16[n*D + k] = (bf16) w_f32[k*D + n]   (tiny, L2-resident after)
__global__ void wt_convert(const float* __restrict__ in, __bf16* __restrict__ out,
                           int D)
{
    int i = blockIdx.x * blockDim.x + threadIdx.x;
    if (i < D * D) {
        int n = i / D, k = i - n * D;
        out[i] = (__bf16)in[k * D + n];
    }
}

extern "C" void kernel_launch(void* const* d_in, const int* in_sizes, int n_in,
                              void* d_out, int out_size, void* d_ws, size_t ws_size,
                              hipStream_t stream)
{
    (void)n_in; (void)out_size; (void)ws_size;
    const int arity[4] = {1, 2, 2, 3};

    const float* x = (const float*)d_in[0];
    const float* wA[4]; const float* bA[4];
    const float* wB[4]; const float* bB[4];
    const int*   idx[4];
    int base = 1;
    for (int r = 0; r < 4; ++r) {
        wA[r]  = (const float*)d_in[base + 0];
        bA[r]  = (const float*)d_in[base + 1];
        wB[r]  = (const float*)d_in[base + 2];
        bB[r]  = (const float*)d_in[base + 3];
        idx[r] = (const int*)  d_in[base + 4];
        base += 5;
    }
    const int nFacts = in_sizes[5];   // idx1 count == N_FACTS (arity 1)

    // bf16 transposed weights in workspace (147456 elems = 288 KB total)
    __bf16* ws = (__bf16*)d_ws;
    __bf16* wAt[4]; __bf16* wBt[4];
    size_t off = 0;
    for (int r = 0; r < 4; ++r) {
        int D = arity[r] * 64;
        wAt[r] = ws + off; off += (size_t)D * D;
        wBt[r] = ws + off; off += (size_t)D * D;
    }
    for (int r = 0; r < 4; ++r) {
        int D = arity[r] * 64, n = D * D;
        wt_convert<<<(n + 255) / 256, 256, 0, stream>>>(wA[r], wAt[r], D);
        wt_convert<<<(n + 255) / 256, 256, 0, stream>>>(wB[r], wBt[r], D);
    }

    float* out = (float*)d_out;
    const int blocks = (nFacts + 31) / 32;
    size_t outOff = 0;

    rel_msg_kernel<1><<<blocks, 128, 0, stream>>>(x, idx[0], wAt[0], bA[0],
                                                  wBt[0], bB[0], out + outOff, nFacts);
    outOff += (size_t)nFacts * 64;
    rel_msg_kernel<2><<<blocks, 128, 0, stream>>>(x, idx[1], wAt[1], bA[1],
                                                  wBt[1], bB[1], out + outOff, nFacts);
    outOff += (size_t)nFacts * 128;
    rel_msg_kernel<2><<<blocks, 128, 0, stream>>>(x, idx[2], wAt[2], bA[2],
                                                  wBt[2], bB[2], out + outOff, nFacts);
    outOff += (size_t)nFacts * 128;
    rel_msg_kernel<3><<<blocks, 128, 0, stream>>>(x, idx[3], wAt[3], bA[3],
                                                  wBt[3], bB[3], out + outOff, nFacts);
    outOff += (size_t)nFacts * 192;

    // out_indices: raw int32 copied into the tail of d_out (capture-safe d2d)
    char* idst = (char*)d_out + outOff * sizeof(float);
    for (int r = 0; r < 4; ++r) {
        size_t bytes = (size_t)nFacts * arity[r] * sizeof(int);
        hipMemcpyAsync(idst, idx[r], bytes, hipMemcpyDeviceToDevice, stream);
        idst += bytes;
    }
}
